// MultiHeadBindingAttention_37288906064324
// MI455X (gfx1250) — matrix operand order repaired and checked by reading, not compiled, not run on hardware
//
#include <hip/hip_runtime.h>
#include <hip/hip_bf16.h>

// ---------------------------------------------------------------------------
// MultiHeadBindingAttention for MI455X (gfx1250, wave32, WMMA bf16)
//
// out[b,t,h,d] = vb[h,d] * sum_{s<=t} sigmoid( sum_k x[b,t,h,k] x[b,s,h,k] qk[h,k] ) * x[b,s,h,d]
//   qk[h,k] = alpha_q[h]*alpha_k[h]*sgn(bv_q)*sgn(bv_k) * (4/sqrt(HD))   (folded)
//   vb[h,d] = alpha_v[h]*sgn(bv_v)
// Sigmoid attention => no softmax running stats; pure masked GEMM pipeline.
//
// Work balance: each workgroup processes the complementary pair of query
// blocks (y, 63-y) => exactly 65 key blocks per workgroup (uniform grid).
// ---------------------------------------------------------------------------

#define Bn   4
#define Tn   2048
#define NH   4
#define HD   1024
#define HDC  4096   // NH*HD
#define QB   (Tn / 32)   // 64 query blocks of 32 rows

typedef __attribute__((ext_vector_type(16))) __bf16 v16bf;
typedef __attribute__((ext_vector_type(8)))  float  v8f;

static __device__ __forceinline__ v8f zero8() {
  v8f z = {0.f, 0.f, 0.f, 0.f, 0.f, 0.f, 0.f, 0.f};
  return z;
}

// ---------------------------------------------------------------------------
// Kernel A: per-head binding vectors. one block per head.
// ---------------------------------------------------------------------------
__global__ __launch_bounds__(256)
void bind_kernel(const float* __restrict__ bvq, const float* __restrict__ bvk,
                 const float* __restrict__ bvv, float* __restrict__ qk,
                 float* __restrict__ vb) {
  __shared__ float rq[256], rk[256], rv[256];
  const int h = blockIdx.x;
  const int t = threadIdx.x;
  const float* qh = bvq + h * HD;
  const float* kh = bvk + h * HD;
  const float* vh = bvv + h * HD;
  float sq = 0.f, sk = 0.f, sv = 0.f;
  for (int d = t; d < HD; d += 256) {
    sq += fabsf(qh[d]); sk += fabsf(kh[d]); sv += fabsf(vh[d]);
  }
  rq[t] = sq; rk[t] = sk; rv[t] = sv;
  __syncthreads();
  for (int off = 128; off > 0; off >>= 1) {
    if (t < off) { rq[t] += rq[t + off]; rk[t] += rk[t + off]; rv[t] += rv[t + off]; }
    __syncthreads();
  }
  const float aq = rq[0] * (1.f / HD);
  const float ak = rk[0] * (1.f / HD);
  const float av = rv[0] * (1.f / HD);
  // fold sigmoid-gain(4) * attention scale(1/sqrt(HD)=1/32) = 0.125 into qk
  const float c = aq * ak * 0.125f;
  for (int d = t; d < HD; d += 256) {
    float gq = (qh[d] < 0.f) ? -1.f : 1.f;   // sign with 0 -> +alpha (STE rule)
    float gk = (kh[d] < 0.f) ? -1.f : 1.f;
    float gv = (vh[d] < 0.f) ? -1.f : 1.f;
    qk[h * HD + d] = c * gq * gk;
    vb[h * HD + d] = av * gv;
  }
}

// ---------------------------------------------------------------------------
// Kernel B1: cast x (f32) -> Xb (bf16), same [B,T,H*HD] layout. 64 MB, L2-resident.
// ---------------------------------------------------------------------------
__global__ __launch_bounds__(256)
void cast_kernel(const float* __restrict__ x, __bf16* __restrict__ xb, size_t n) {
  size_t i = ((size_t)blockIdx.x * 256 + threadIdx.x) * 4;
  if (i + 3 < n) {
    float4 f = *(const float4*)(x + i);
    xb[i + 0] = (__bf16)f.x;
    xb[i + 1] = (__bf16)f.y;
    xb[i + 2] = (__bf16)f.z;
    xb[i + 3] = (__bf16)f.w;
  }
}

// ---------------------------------------------------------------------------
// Kernel B2: transposed bf16 copy XbT[(b,h)][d][s] for the PV GEMM B-operand
// (keys become the contiguous axis => B-fragment loads are 32B contiguous).
// ---------------------------------------------------------------------------
__global__ __launch_bounds__(256)
void transpose_kernel(const float* __restrict__ x, __bf16* __restrict__ xbt) {
  __shared__ float tile[32][33];
  const int bh = blockIdx.z;           // b*NH + h
  const int b  = bh >> 2;
  const int h  = bh & 3;
  const int d0 = blockIdx.x * 32;
  const int s0 = blockIdx.y * 32;
  const int tx = threadIdx.x;          // 0..31
  const int ty = threadIdx.y;          // 0..7
#pragma unroll
  for (int i = 0; i < 4; ++i) {
    int s = s0 + ty + 8 * i;
    tile[ty + 8 * i][tx] = x[((size_t)b * Tn + s) * HDC + h * HD + d0 + tx];
  }
  __syncthreads();
#pragma unroll
  for (int i = 0; i < 4; ++i) {
    int d = d0 + ty + 8 * i;
    xbt[((size_t)bh * HD + d) * Tn + s0 + tx] = (__bf16)tile[tx][ty + 8 * i];
  }
}

// ---------------------------------------------------------------------------
// Kernel C: fused sigmoid-attention.
// grid = (B*NH, T/64), block = 256 (8 waves).
// Wave w owns feature slice [128w, 128w+128): K-slice of the score GEMM and
// N-slice of the output GEMM. Partial 32x32 S tiles are reduced across waves
// with ds_add_f32 into a double-buffered LDS tile.
//
// Pipelining: ALL global B-fragment loads for a key block (S-phase Bs[2][4]
// and PV-phase Bt[8]) are issued into distinct registers before the first
// barrier; the barrier + reduction phase hides their latency and the compiler
// retires them with partial s_wait_loadcnt interleaved with WMMAs.
// ---------------------------------------------------------------------------
__global__ __launch_bounds__(256)
void attn_kernel(const float* __restrict__ x, const float* __restrict__ qk,
                 const float* __restrict__ vb, const __bf16* __restrict__ Xb,
                 const __bf16* __restrict__ XbT, float* __restrict__ out) {
  __shared__ float Sbuf[2][1024];   // two 32x32 f32 score tiles (double buffer)

  const int bh = blockIdx.x;
  const int b  = bh >> 2;
  const int h  = bh & 3;

  const int wave   = threadIdx.x >> 5;   // 0..7
  const int lane   = threadIdx.x & 31;
  const int g      = lane >> 4;          // K-half selector (A/B layout)
  const int m      = lane & 15;          // row (A) / col (B,C,D) within tile
  const int kchunk = wave * 128;         // this wave's feature slice

  // loop-invariant fragment base pointers
  const __bf16* xs_base = Xb  + ((size_t)b * Tn + m) * HDC + h * HD + kchunk + 16 * g;
  const __bf16* xt_base = XbT + ((size_t)bh * HD + kchunk + m) * Tn + 16 * g;
  const float*  qkr     = qk + h * HD;
  const float*  vbr     = vb + h * HD;

  // complementary pair of query blocks: uniform 65 key blocks per workgroup
#pragma unroll 1
  for (int rep = 0; rep < 2; ++rep) {
    const int qb = rep ? (QB - 1 - blockIdx.y) : blockIdx.y;
    const int t0 = qb * 32;

    // ---- build Q A-fragments: Q = x * qk (bf16), 2 Mtiles x 4 Ksteps
    v16bf qA[2][4];
#pragma unroll
    for (int mt = 0; mt < 2; ++mt) {
      const int row = t0 + 16 * mt + m;
      const float* xr = x + ((size_t)b * Tn + row) * HDC + h * HD;
#pragma unroll
      for (int ks = 0; ks < 4; ++ks) {
        const int base = kchunk + 32 * ks + 16 * g;
#pragma unroll
        for (int j = 0; j < 8; ++j) {
          qA[mt][ks][j]     = (__bf16)(xr[base + j]      * qkr[base + j]);
          qA[mt][ks][8 + j] = (__bf16)(xr[base + 8 + j] * qkr[base + 8 + j]);
        }
      }
    }

    // ---- output accumulators: 2 Mtiles x 8 Ntiles (16 feats) = 32x128 f32
    v8f Oacc[2][8];
#pragma unroll
    for (int mt = 0; mt < 2; ++mt)
#pragma unroll
      for (int nt = 0; nt < 8; ++nt) Oacc[mt][nt] = zero8();

    // ---- stream causal key blocks of 32
    for (int kb = 0; kb <= qb; ++kb) {
      float* Sb = Sbuf[kb & 1];
      {
        float4 z4 = make_float4(0.f, 0.f, 0.f, 0.f);
        ((float4*)Sb)[threadIdx.x] = z4;   // 256 threads x 16B = 4KB
      }

      // ---- issue ALL global loads for this key block up front ----
      v16bf Bs[2][4];                       // S-phase  B: keys x feature-chunk
      const __bf16* xs = xs_base + (size_t)(kb * 32) * HDC;
#pragma unroll
      for (int nt = 0; nt < 2; ++nt) {
        const __bf16* xr = xs + (size_t)(16 * nt) * HDC;
#pragma unroll
        for (int ks = 0; ks < 4; ++ks)
          Bs[nt][ks] = *(const v16bf*)(xr + 32 * ks);
      }
      v16bf Bt[8];                          // PV-phase B: feature x keys
      const __bf16* xt = xt_base + kb * 32;
#pragma unroll
      for (int nt = 0; nt < 8; ++nt)
        Bt[nt] = *(const v16bf*)(xt + (size_t)(16 * nt) * Tn);

      // prefetch next key block (global_prefetch_b8)
      if (kb < qb) {
        __builtin_prefetch(xs + (size_t)32 * HDC + (size_t)(16 * g) * HDC, 0, 1);
        __builtin_prefetch(xt + 32 + (size_t)(16 * g) * Tn, 0, 1);
      }

      __syncthreads();   // zero visible; load latency hidden behind barrier

      // -- partial scores over this wave's 128-feature K-slice (16 WMMAs)
      v8f sacc[2][2];
#pragma unroll
      for (int mt = 0; mt < 2; ++mt)
#pragma unroll
        for (int nt = 0; nt < 2; ++nt) sacc[mt][nt] = zero8();

#pragma unroll
      for (int nt = 0; nt < 2; ++nt)
#pragma unroll
        for (int ks = 0; ks < 4; ++ks)
#pragma unroll
          for (int mt = 0; mt < 2; ++mt)
            sacc[mt][nt] = __builtin_amdgcn_wmma_f32_16x16x32_bf16(
                false, qA[mt][ks], false, Bs[nt][ks], (short)0, sacc[mt][nt], false, false);

      // -- cross-wave K reduction via LDS float atomics (ds_add_f32)
#pragma unroll
      for (int mt = 0; mt < 2; ++mt)
#pragma unroll
        for (int nt = 0; nt < 2; ++nt)
#pragma unroll
          for (int r = 0; r < 8; ++r)
            atomicAdd(&Sb[(16 * mt + 8 * g + r) * 32 + 16 * nt + m], sacc[mt][nt][r]);
      __syncthreads();

      // -- P = causal-masked sigmoid(S), packed as bf16 A-fragments (16x32)
      v16bf pA[2];
#pragma unroll
      for (int mt = 0; mt < 2; ++mt) {
        const int row = t0 + 16 * mt + m;
        const float* srow = &Sb[(16 * mt + m) * 32];
#pragma unroll
        for (int e = 0; e < 16; ++e) {
          const int kk  = e + 16 * g;
          const int key = kb * 32 + kk;
          const float sv = srow[kk];
          const float p  = (key <= row) ? (1.0f / (1.0f + __expf(-sv))) : 0.0f;
          pA[mt][e] = (__bf16)p;
        }
      }

      // -- O += P @ X : Bt fragments were fetched a whole phase ago (16 WMMAs)
#pragma unroll
      for (int nt = 0; nt < 8; ++nt)
#pragma unroll
        for (int mt = 0; mt < 2; ++mt)
          Oacc[mt][nt] = __builtin_amdgcn_wmma_f32_16x16x32_bf16(
              false, pA[mt], false, Bt[nt], (short)0, Oacc[mt][nt], false, false);
    }

    // ---- epilogue: out = Oacc * vb   (C/D layout: M = r + 8*g, N = lane%16)
#pragma unroll
    for (int mt = 0; mt < 2; ++mt) {
#pragma unroll
      for (int nt = 0; nt < 8; ++nt) {
        const int feat = kchunk + 16 * nt + m;
        const float vbv = vbr[feat];
#pragma unroll
        for (int r = 0; r < 8; ++r) {
          const int row = t0 + 16 * mt + 8 * g + r;
          out[((size_t)b * Tn + row) * HDC + h * HD + feat] = Oacc[mt][nt][r] * vbv;
        }
      }
    }

    // rep boundary: restore the two-barrier distance for the double buffer
    __syncthreads();
  }
}

// ---------------------------------------------------------------------------
extern "C" void kernel_launch(void* const* d_in, const int* in_sizes, int n_in,
                              void* d_out, int out_size, void* d_ws, size_t ws_size,
                              hipStream_t stream) {
  (void)in_sizes; (void)n_in; (void)out_size; (void)ws_size;
  const float* x   = (const float*)d_in[0];
  const float* bvq = (const float*)d_in[1];
  const float* bvk = (const float*)d_in[2];
  const float* bvv = (const float*)d_in[3];
  float* out = (float*)d_out;

  // workspace layout: [qk 16KB][vb 16KB][pad to 64KB][Xb 64MB][XbT 64MB]
  float*  qk  = (float*)d_ws;
  float*  vb  = qk + NH * HD;
  __bf16* Xb  = (__bf16*)((char*)d_ws + (64 << 10));
  __bf16* XbT = Xb + (size_t)Bn * Tn * HDC;

  bind_kernel<<<NH, 256, 0, stream>>>(bvq, bvk, bvv, qk, vb);

  const size_t n = (size_t)Bn * Tn * HDC;
  cast_kernel<<<(unsigned)((n / 4 + 255) / 256), 256, 0, stream>>>(x, Xb, n);

  dim3 tgrid(HD / 32, Tn / 32, Bn * NH);
  transpose_kernel<<<tgrid, dim3(32, 8), 0, stream>>>(x, XbT);

  dim3 agrid(Bn * NH, QB / 2);
  attn_kernel<<<agrid, 256, 0, stream>>>(x, qk, vb, Xb, XbT, out);
}
